// LoRAMulti_MLP_38302518346465
// MI455X (gfx1250) — matrix operand order repaired and checked by reading, not compiled, not run on hardware
//
#include <hip/hip_runtime.h>

#define T_TASKS 8
#define BATCH 1024
#define LRANK 8
#define SCALING 2.0f

typedef float v8f __attribute__((ext_vector_type(8)));
typedef __bf16 v16bf __attribute__((ext_vector_type(16)));

struct __align__(16) U4 { unsigned int x, y, z, w; };
union Frag { v16bf v; U4 q[2]; };

__device__ __forceinline__ unsigned short f2bf(float f) {
  unsigned int u = __float_as_uint(f);
  u += 0x7FFFu + ((u >> 16) & 1u);   // round-to-nearest-even
  return (unsigned short)(u >> 16);
}
__device__ __forceinline__ float bf2f(unsigned short h) {
  return __uint_as_float(((unsigned int)h) << 16);
}

__global__ __launch_bounds__(256) void cvt_f32_bf16(const float* __restrict__ in,
                                                    unsigned short* __restrict__ out,
                                                    int n) {
  for (int i = blockIdx.x * 256 + threadIdx.x; i < n; i += gridDim.x * 256)
    out[i] = f2bf(in[i]);
}

// one-time: k[K][H] f32 -> kT[H][K] bf16 (32x32 LDS-tiled transpose)
__global__ __launch_bounds__(256) void cvt_transpose_bf16(const float* __restrict__ in,
                                                          unsigned short* __restrict__ out,
                                                          int K, int H) {
  __shared__ unsigned short tile[32][33];
  const int hb = blockIdx.x * 32, kb = blockIdx.y * 32;
  const int tx = threadIdx.x & 31, ty = threadIdx.x >> 5;  // ty: 0..7
#pragma unroll
  for (int i = 0; i < 32; i += 8)
    tile[ty + i][tx] = f2bf(in[(size_t)(kb + ty + i) * H + hb + tx]);
  __syncthreads();
#pragma unroll
  for (int i = 0; i < 32; i += 8)
    out[(size_t)(hb + ty + i) * K + kb + tx] = tile[tx][ty + i];
}

// a[t,b,r] = sum_i h[t,b,i] * d[i,r,t]   (rank-8 down projection, tiny)
__global__ __launch_bounds__(256) void lora_down(const unsigned short* __restrict__ hb,
                                                 const float* __restrict__ d,
                                                 float* __restrict__ a, int K) {
  int idx = blockIdx.x * 256 + threadIdx.x;   // T*B*8 threads
  int r = idx & 7;
  int b = (idx >> 3) & (BATCH - 1);
  int t = idx >> 13;
  const unsigned short* hrow = hb + (size_t)(t * BATCH + b) * K;
  const float* dcol = d + (size_t)r * T_TASKS + t;
  float s = 0.f;
  for (int i = 0; i < K; i += 8) {
    U4 qq = *(const U4*)(hrow + i);
    const unsigned short* e = (const unsigned short*)&qq;
#pragma unroll
    for (int j = 0; j < 8; ++j)
      s += bf2f(e[j]) * dcol[(size_t)(i + j) * (LRANK * T_TASKS)];
  }
  a[idx] = s;
}

// out[t, bm+0..127, bn+0..127] = act( h @ k + bias + SCALING * (a @ u) )
template <int ACT, int STORE_F32>
__global__ __launch_bounds__(256) void lora_gemm(
    const unsigned short* __restrict__ hb,   // [T*B, K] bf16
    const unsigned short* __restrict__ kt,   // [H, K]   bf16 (pre-transposed)
    const float* __restrict__ bias,          // [H]
    const float* __restrict__ av,            // [T*B, 8]
    const float* __restrict__ uv,            // [8, H, T]
    float* __restrict__ out32,               // [T*B, H] (final layer)
    unsigned short* __restrict__ out16,      // [T*B, H] bf16 (intermediate)
    int K, int H) {
  __shared__ __align__(16) unsigned short sA[128 * 40];  // A tile [m][k], pad 40
  __shared__ __align__(16) unsigned short sB[128 * 40];  // B tile [n][k], pad 40
  __shared__ float sLa[128 * 8];                         // a rows for this m-tile
  __shared__ float sU[8 * 128];                          // u cols for this n-tile
  __shared__ float sBias[128];

  const int tid = threadIdx.x;
  const int wave = tid >> 5, lane = tid & 31;
  const int l15 = lane & 15;
  const int half = lane >> 4;                // 0: lanes 0-15, 1: lanes 16-31
  const int t = blockIdx.z;
  const int bn = blockIdx.x * 128;
  const int bm = blockIdx.y * 128;
  const int mw = (wave >> 1) * 32;           // 4 M-strips of 32
  const int nw = (wave & 1) * 64;            // 2 N-strips of 64

  // preload LoRA a-rows, u-cols, bias (covered by first barrier in k-loop)
  for (int s = tid; s < 128 * 8; s += 256)
    sLa[s] = av[((size_t)(t * BATCH + bm) << 3) + s];
  for (int s = tid; s < 8 * 128; s += 256) {
    int r = s >> 7, n = s & 127;
    sU[s] = uv[((size_t)r * H + bn + n) * T_TASKS + t];
  }
  if (tid < 128) sBias[tid] = bias[bn + tid];

  // staging slots: 512 slots of 8 bf16 (16B) each for A and for B; 2 per thread
  const int rowA0 = tid >> 2, chA0 = tid & 3;
  const int rowA1 = (tid + 256) >> 2, chA1 = tid & 3;  // (tid+256)&3 == tid&3
  const unsigned short* pA0 = hb + (size_t)(t * BATCH + bm + rowA0) * K + chA0 * 8;
  const unsigned short* pA1 = hb + (size_t)(t * BATCH + bm + rowA1) * K + chA1 * 8;
  const unsigned short* pB0 = kt + (size_t)(bn + rowA0) * K + chA0 * 8;
  const unsigned short* pB1 = kt + (size_t)(bn + rowA1) * K + chA1 * 8;
  unsigned short* dA0 = &sA[rowA0 * 40 + chA0 * 8];
  unsigned short* dA1 = &sA[rowA1 * 40 + chA1 * 8];
  unsigned short* dB0 = &sB[rowA0 * 40 + chA0 * 8];
  unsigned short* dB1 = &sB[rowA1 * 40 + chA1 * 8];

  // fragment LDS addresses (fixed per lane)
  const unsigned short* fA0 = &sA[(mw + l15) * 40 + half * 8];
  const unsigned short* fA1 = &sA[(mw + 16 + l15) * 40 + half * 8];
  const unsigned short* fB[4] = {
      &sB[(nw + 0 + l15) * 40 + half * 8], &sB[(nw + 16 + l15) * 40 + half * 8],
      &sB[(nw + 32 + l15) * 40 + half * 8], &sB[(nw + 48 + l15) * 40 + half * 8]};

  v8f acc[2][4] = {};

  for (int k0 = 0; k0 < K; k0 += 32) {
    // issue all 4 global loads up front (overlap with previous iter's WMMA)
    U4 ra0 = *(const U4*)pA0;
    U4 ra1 = *(const U4*)pA1;
    U4 rb0 = *(const U4*)pB0;
    U4 rb1 = *(const U4*)pB1;
    // unconditional speculative prefetch of next K-tile (dropped if OOB)
    __builtin_prefetch(pA0 + 32, 0, 0);
    __builtin_prefetch(pB0 + 32, 0, 0);
    pA0 += 32; pA1 += 32; pB0 += 32; pB1 += 32;

    __syncthreads();   // previous iteration's fragment reads complete
    *(U4*)dA0 = ra0;
    *(U4*)dA1 = ra1;
    *(U4*)dB0 = rb0;
    *(U4*)dB1 = rb1;
    __syncthreads();   // tiles visible

    // build fragments per CDNA5 16-bit WMMA lane layout
    Frag fa[2], fb[4];
    fa[0].q[0] = *(const U4*)fA0;            // K {0..7 | 8..15}
    fa[0].q[1] = *(const U4*)(fA0 + 16);     // K {16..23 | 24..31}
    fa[1].q[0] = *(const U4*)fA1;
    fa[1].q[1] = *(const U4*)(fA1 + 16);
#pragma unroll
    for (int ns = 0; ns < 4; ++ns) {
      fb[ns].q[0] = *(const U4*)fB[ns];      // K {0..7 | 8..15}
      fb[ns].q[1] = *(const U4*)(fB[ns] + 16);
    }
#pragma unroll
    for (int ms = 0; ms < 2; ++ms)
#pragma unroll
      for (int ns = 0; ns < 4; ++ns)
        acc[ms][ns] = __builtin_amdgcn_wmma_f32_16x16x32_bf16(
            false, fa[ms].v, false, fb[ns].v, (short)0, acc[ms][ns], false, false);
  }

  // epilogue: bias + SCALING*(a@u) + ReLU, store
#pragma unroll
  for (int ms = 0; ms < 2; ++ms) {
#pragma unroll
    for (int ns = 0; ns < 4; ++ns) {
#pragma unroll
      for (int v = 0; v < 8; ++v) {
        int mi = mw + ms * 16 + v + half * 8;   // C layout: vgpr v -> M=v (+8 hi lanes)
        int ni = nw + ns * 16 + l15;
        float val = acc[ms][ns][v] + sBias[ni];
        float dsum = 0.f;
#pragma unroll
        for (int r = 0; r < 8; ++r) dsum += sLa[mi * 8 + r] * sU[r * 128 + ni];
        val += SCALING * dsum;
        if (ACT) val = fmaxf(val, 0.f);
        size_t gi = (size_t)(t * BATCH + bm + mi) * H + bn + ni;
        if (STORE_F32) out32[gi] = val;
        else out16[gi] = f2bf(val);
      }
    }
  }
}

extern "C" void kernel_launch(void* const* d_in, const int* in_sizes, int n_in,
                              void* d_out, int out_size, void* d_ws, size_t ws_size,
                              hipStream_t stream) {
  (void)in_sizes; (void)n_in; (void)out_size; (void)ws_size;
  const float* x  = (const float*)d_in[0];
  const float* k0 = (const float*)d_in[1];
  const float* b0 = (const float*)d_in[2];
  const float* d0 = (const float*)d_in[3];
  const float* u0 = (const float*)d_in[4];
  const float* k1 = (const float*)d_in[5];
  const float* b1 = (const float*)d_in[6];
  const float* d1 = (const float*)d_in[7];
  const float* u1 = (const float*)d_in[8];
  const float* k2 = (const float*)d_in[9];
  const float* b2 = (const float*)d_in[10];
  const float* d2 = (const float*)d_in[11];
  const float* u2 = (const float*)d_in[12];

  const int T = 8, B = 1024, D = 1024, H1 = 2048, H2 = 2048, H3 = 1024;

  char* ws = (char*)d_ws;
  size_t off = 0;
  auto alloc = [&](size_t bytes) {
    size_t o = off; off += (bytes + 255) & ~(size_t)255; return o;
  };
  unsigned short* xb  = (unsigned short*)(ws + alloc((size_t)T * B * D * 2));
  unsigned short* h1b = (unsigned short*)(ws + alloc((size_t)T * B * H1 * 2));
  unsigned short* h2b = (unsigned short*)(ws + alloc((size_t)T * B * H2 * 2));
  unsigned short* k0t = (unsigned short*)(ws + alloc((size_t)D * H1 * 2));
  unsigned short* k1t = (unsigned short*)(ws + alloc((size_t)H1 * H2 * 2));
  unsigned short* k2t = (unsigned short*)(ws + alloc((size_t)H2 * H3 * 2));
  float* av = (float*)(ws + alloc((size_t)T * B * 8 * 4));

  cvt_f32_bf16<<<2048, 256, 0, stream>>>(x, xb, T * B * D);
  cvt_transpose_bf16<<<dim3(H1 / 32, D / 32), 256, 0, stream>>>(k0, k0t, D, H1);
  cvt_transpose_bf16<<<dim3(H2 / 32, H1 / 32), 256, 0, stream>>>(k1, k1t, H1, H2);
  cvt_transpose_bf16<<<dim3(H3 / 32, H2 / 32), 256, 0, stream>>>(k2, k2t, H2, H3);

  // layer 0: h1 = relu(x @ k0 + b0 + s*(x@d0)@u0)
  lora_down<<<T * B * 8 / 256, 256, 0, stream>>>(xb, d0, av, D);
  lora_gemm<1, 0><<<dim3(H1 / 128, B / 128, T), 256, 0, stream>>>(
      xb, k0t, b0, av, u0, nullptr, h1b, D, H1);
  // layer 1
  lora_down<<<T * B * 8 / 256, 256, 0, stream>>>(h1b, d1, av, H1);
  lora_gemm<1, 0><<<dim3(H2 / 128, B / 128, T), 256, 0, stream>>>(
      h1b, k1t, b1, av, u1, nullptr, h2b, H1, H2);
  // layer 2 (no activation, f32 output)
  lora_down<<<T * B * 8 / 256, 256, 0, stream>>>(h2b, d2, av, H2);
  lora_gemm<0, 1><<<dim3(H3 / 128, B / 128, T), 256, 0, stream>>>(
      h2b, k2t, b2, av, u2, (float*)d_out, nullptr, H2, H3);
}
